// PAM_57552561766474
// MI455X (gfx1250) — matrix operand order repaired and checked by reading, not compiled, not run on hardware
//
#include <hip/hip_runtime.h>
#include <hip/hip_bf16.h>

// ---- CDNA5 WMMA vector types ----
typedef __attribute__((ext_vector_type(16))) __bf16 v16bf;
typedef __attribute__((ext_vector_type(8)))  float  v8f;

#define BATCH 4
#define NTOK  4096   // H*W
#define CHN   512    // C
#define DQK   64     // C/8

union V16 { uint4 q[2]; unsigned int u[8]; v16bf v; };

static __device__ __forceinline__ unsigned short f2bf(float f) {
  union { float f; unsigned int u; } c; c.f = f;
  unsigned int r = c.u + 0x7FFFu + ((c.u >> 16) & 1u);  // RNE-ish
  return (unsigned short)(r >> 16);
}

// packed f32x2 -> bf16x2 (single packed convert when available)
static __device__ __forceinline__ unsigned int pk2(float lo, float hi) {
#if __has_builtin(__builtin_amdgcn_cvt_pk_bf16_f32)
  typedef __attribute__((ext_vector_type(2))) __bf16 v2bf;
  union { v2bf v; unsigned int u; } c;
  c.v = __builtin_amdgcn_cvt_pk_bf16_f32(lo, hi);
  return c.u;
#else
  return (unsigned int)f2bf(lo) | ((unsigned int)f2bf(hi) << 16);
#endif
}

static __device__ __forceinline__ float fast_rcp(float y) {
#if __has_builtin(__builtin_amdgcn_rcpf)
  return __builtin_amdgcn_rcpf(y);     // v_rcp_f32
#else
  return 1.0f / y;
#endif
}

// ---------------------------------------------------------------------------
// Kernel 0: transpose + convert weights into one unified bf16 buffer.
// WT rows 0-63 = Wq^T, 64-127 = Wk^T, 128-639 = Wv^T (row = out-col, col = c)
// ---------------------------------------------------------------------------
__global__ void pam_wt(const float* __restrict__ Wq, const float* __restrict__ Wk,
                       const float* __restrict__ Wv, unsigned short* __restrict__ WT) {
  int idx = blockIdx.x * blockDim.x + threadIdx.x;     // 640*512 total
  if (idx >= 640 * CHN) return;
  int row = idx >> 9;
  int c   = idx & (CHN - 1);
  float v;
  if (row < 64)        v = Wq[c * DQK + row];
  else if (row < 128)  v = Wk[c * DQK + (row - 64)];
  else                 v = Wv[c * CHN + (row - 128)];
  WT[idx] = f2bf(v);
}

// ---------------------------------------------------------------------------
// Kernel 1: fused projection GEMM [16384,512] x [512,640] (bf16 WMMA),
// W tiles double-buffered through LDS; outputs q,k row-major + v transposed.
// ---------------------------------------------------------------------------
__global__ __launch_bounds__(256, 1) void pam_proj(
    const float* __restrict__ x, const unsigned short* __restrict__ WT,
    unsigned short* __restrict__ qws, unsigned short* __restrict__ kws,
    unsigned short* __restrict__ vT) {
  __shared__ __align__(16) unsigned short w_lds[2][640 * 32];   // 2 x 40KB

  const int tid  = threadIdx.x;
  const int b    = blockIdx.x >> 6;
  const int row0 = (blockIdx.x & 63) * 64;
  const int w    = tid >> 5;
  const int lane = tid & 31;
  const int h    = lane >> 4, l16 = lane & 15;
  const int mblk = w >> 1, colhalf = w & 1;

  const int n = row0 + mblk * 16 + l16;                 // this lane's A row
  const float* xr = x + ((size_t)(b * NTOK + n)) * CHN;

  v8f z = {};
  v8f acc[20];
#pragma unroll
  for (int i = 0; i < 20; ++i) acc[i] = z;

  // --- W tile staging: 640 rows x 32 K bf16 = 2560 uint4, 10 per thread ---
  uint4 wst[10];
#define ISSUE_W(ks)                                                            \
  {                                                                            \
    _Pragma("unroll") for (int r = 0; r < 10; ++r) {                           \
      int idx = r * 256 + tid;                                                 \
      int row = idx >> 2, part = idx & 3;                                      \
      wst[r] = *(const uint4*)(WT + (size_t)row * CHN + (ks) * 32 + part * 8); \
    }                                                                          \
  }
#define COMMIT_W(p)                                                            \
  {                                                                            \
    _Pragma("unroll") for (int r = 0; r < 10; ++r)                             \
        ((uint4*)w_lds[p])[r * 256 + tid] = wst[r];                            \
  }

  ISSUE_W(0);

#pragma unroll 1
  for (int ks = 0; ks < 16; ++ks) {                     // K = 512 in steps of 32
    const int p = ks & 1;
    // A-tile source loads (issue early; independent of LDS)
    const int c0 = ks * 32 + 16 * h;
    float4 f0 = *(const float4*)(xr + c0);
    float4 f1 = *(const float4*)(xr + c0 + 4);
    float4 f2 = *(const float4*)(xr + c0 + 8);
    float4 f3 = *(const float4*)(xr + c0 + 12);

    COMMIT_W(p);
    __syncthreads();
    if (ks + 1 < 16) ISSUE_W(ks + 1);                   // prefetch next W tile

    V16 a;                                              // A 16x32: VGPRi K=16h+2i
    a.u[0] = pk2(f0.x, f0.y); a.u[1] = pk2(f0.z, f0.w);
    a.u[2] = pk2(f1.x, f1.y); a.u[3] = pk2(f1.z, f1.w);
    a.u[4] = pk2(f2.x, f2.y); a.u[5] = pk2(f2.z, f2.w);
    a.u[6] = pk2(f3.x, f3.y); a.u[7] = pk2(f3.z, f3.w);

    const unsigned short* wl = w_lds[p];
    // B-tile 3-deep rotating pipeline so ds_loads stay ahead of WMMAs
    V16 bt[3];
#pragma unroll
    for (int pr = 0; pr < 2; ++pr) {
      const uint4* wp = (const uint4*)(wl + ((colhalf * 20 + pr) * 16 + l16) * 32 + 16 * h);
      bt[pr].q[0] = wp[0]; bt[pr].q[1] = wp[1];
    }
#pragma unroll
    for (int cb = 0; cb < 20; ++cb) {
      if (cb + 2 < 20) {
        const int cbg = colhalf * 20 + cb + 2;
        const uint4* wp = (const uint4*)(wl + (cbg * 16 + l16) * 32 + 16 * h);
        bt[(cb + 2) % 3].q[0] = wp[0]; bt[(cb + 2) % 3].q[1] = wp[1];
      }
      acc[cb] = __builtin_amdgcn_wmma_f32_16x16x32_bf16(
          false, a.v, false, bt[cb % 3].v, (short)0, acc[cb], false, false);
    }
  }

  // Epilogue: C/D layout VGPR i -> M = 8h+i, N = l16
  const int nbase = row0 + mblk * 16 + 8 * h;
#pragma unroll
  for (int cb = 0; cb < 20; ++cb) {
    const int cbg = colhalf * 20 + cb;
    if (cbg < 8) {                                      // q or k: [b][n][64] bf16
      unsigned short* dst = (cbg < 4) ? qws : kws;
      const int d = (cbg & 3) * 16 + l16;
#pragma unroll
      for (int i = 0; i < 8; ++i)
        dst[((size_t)(b * NTOK + nbase + i)) * DQK + d] = f2bf(acc[cb][i]);
    } else {                                            // v transposed: [b][c][n] bf16
      const int c = (cbg - 8) * 16 + l16;
      unsigned short* dst = vT + ((size_t)(b * CHN + c)) * NTOK + nbase;
#pragma unroll
      for (int j = 0; j < 4; ++j)
        *(unsigned int*)(dst + 2 * j) = pk2(acc[cb][2 * j], acc[cb][2 * j + 1]);
    }
  }
#undef ISSUE_W
#undef COMMIT_W
}

// ---------------------------------------------------------------------------
// Kernel 2: fused sigmoid-attention, double-buffered k/v tiles.
//   acc[n][c] = sum_m sigmoid(q[n].k[m]) * v[m][c];  out = gamma*acc + x
// ---------------------------------------------------------------------------
__global__ __launch_bounds__(256, 1) void pam_attn(
    const float* __restrict__ x, const unsigned short* __restrict__ qws,
    const unsigned short* __restrict__ kws, const unsigned short* __restrict__ vT,
    const float* __restrict__ gamma, float* __restrict__ out) {
  __shared__ __align__(16) unsigned short k_lds[2][32 * 64];     // 2 x 4KB  [key][d]
  __shared__ __align__(16) unsigned short v_lds[2][512 * 32];    // 2 x 32KB [c][m]
  __shared__ __align__(16) unsigned short a_lds[8 * 16 * 32];    // per-wave attn scratch

  const int b   = blockIdx.x >> 6;
  const int q0  = (blockIdx.x & 63) * 64;
  const int tid = threadIdx.x;
  const int w   = tid >> 5, lane = tid & 31;
  const int h   = lane >> 4, l16 = lane & 15;
  const int mblk = w >> 1, chalf = w & 1;

  const unsigned short* kbase = kws + (size_t)b * NTOK * DQK;
  const unsigned short* vbase = vT + (size_t)b * CHN * NTOK;

  // q A-tiles (K = D = 64 -> two 16x32 tiles), kept in registers all kernel
  const int n_q = q0 + mblk * 16 + l16;
  const unsigned short* qr = qws + ((size_t)(b * NTOK + n_q)) * DQK;
  V16 qa[2];
#pragma unroll
  for (int ks = 0; ks < 2; ++ks) {
    qa[ks].q[0] = *(const uint4*)(qr + ks * 32 + 16 * h);
    qa[ks].q[1] = *(const uint4*)(qr + ks * 32 + 16 * h + 8);
  }

  v8f z = {};
  v8f acc[16];
#pragma unroll
  for (int i = 0; i < 16; ++i) acc[i] = z;

  // ---- global->reg staging for tile t (1 uint4 of k, 8 uint4 of v) ----
  const int krow = tid >> 3, kpart = tid & 7;
  uint4 kst;
  uint4 vst[8];
#define ISSUE_T(t)                                                             \
  {                                                                            \
    kst = *(const uint4*)(kbase + (size_t)((t) * 32 + krow) * DQK + kpart * 8);\
    _Pragma("unroll") for (int r = 0; r < 8; ++r) {                            \
      int idx = r * 256 + tid;                                                 \
      int c = idx >> 2, part = idx & 3;                                        \
      vst[r] = *(const uint4*)(vbase + (size_t)c * NTOK + (t) * 32 + part * 8);\
    }                                                                          \
  }
#define COMMIT_T(p)                                                            \
  {                                                                            \
    ((uint4*)k_lds[p])[tid] = kst;                                             \
    _Pragma("unroll") for (int r = 0; r < 8; ++r)                              \
        ((uint4*)v_lds[p])[r * 256 + tid] = vst[r];                            \
  }

  ISSUE_T(0);

#pragma unroll 1
  for (int t = 0; t < 128; ++t) {
    const int p = t & 1;
    COMMIT_T(p);
    __syncthreads();
    if (t + 1 < 128) ISSUE_T(t + 1);                    // prefetch next tile

    const unsigned short* kl = k_lds[p];
    const unsigned short* vl = v_lds[p];

    // scores S[16 x 32] = q @ k^T, sigmoid -> bf16 A layout via LDS scratch
#pragma unroll
    for (int nb = 0; nb < 2; ++nb) {
      const unsigned short* krowp = kl + (nb * 16 + l16) * 64;   // N = key
      V16 kb0, kb1;
      kb0.q[0] = *(const uint4*)(krowp + 16 * h);                // ks=0: K(d)=16h+2i
      kb0.q[1] = *(const uint4*)(krowp + 16 * h + 8);
      kb1.q[0] = *(const uint4*)(krowp + 32 + 16 * h);           // ks=1
      kb1.q[1] = *(const uint4*)(krowp + 32 + 16 * h + 8);
      v8f s = z;
      s = __builtin_amdgcn_wmma_f32_16x16x32_bf16(false, qa[0].v, false, kb0.v, (short)0, s, false, false);
      s = __builtin_amdgcn_wmma_f32_16x16x32_bf16(false, qa[1].v, false, kb1.v, (short)0, s, false, false);
#pragma unroll
      for (int i = 0; i < 8; ++i) {                              // C/D: VGPR i -> M=8h+i
        float sv = fast_rcp(1.0f + __expf(-s[i]));               // sigmoid
        a_lds[(w * 16 + 8 * h + i) * 32 + nb * 16 + l16] = f2bf(sv);
      }
    }

    // reload attn as 16x32 A-matrix (lane = M, packed K pairs)
    V16 at;
    const unsigned short* arow = a_lds + (w * 16 + l16) * 32;
    at.q[0] = *(const uint4*)(arow + 16 * h);
    at.q[1] = *(const uint4*)(arow + 16 * h + 8);

    // acc += attn @ v_tile, B-tile 4-deep rotating pipeline
    V16 vb[4];
#pragma unroll
    for (int pr = 0; pr < 3; ++pr) {
      const unsigned short* vrow = vl + (chalf * 256 + pr * 16 + l16) * 32;
      vb[pr].q[0] = *(const uint4*)(vrow + 16 * h);
      vb[pr].q[1] = *(const uint4*)(vrow + 16 * h + 8);
    }
#pragma unroll
    for (int cb = 0; cb < 16; ++cb) {
      if (cb + 3 < 16) {
        const unsigned short* vrow = vl + (chalf * 256 + (cb + 3) * 16 + l16) * 32;
        vb[(cb + 3) & 3].q[0] = *(const uint4*)(vrow + 16 * h);
        vb[(cb + 3) & 3].q[1] = *(const uint4*)(vrow + 16 * h + 8);
      }
      acc[cb] = __builtin_amdgcn_wmma_f32_16x16x32_bf16(
          false, at.v, false, vb[cb & 3].v, (short)0, acc[cb], false, false);
    }
  }

  // Epilogue: out = gamma*acc + x
  const float g = gamma[0];
#pragma unroll
  for (int cb = 0; cb < 16; ++cb) {
    const int ch = chalf * 256 + cb * 16 + l16;
#pragma unroll
    for (int i = 0; i < 8; ++i) {
      const int n = q0 + mblk * 16 + 8 * h + i;
      const size_t idx = ((size_t)(b * NTOK + n)) * CHN + ch;
      out[idx] = g * acc[cb][i] + x[idx];
    }
  }
#undef ISSUE_T
#undef COMMIT_T
}

// ---------------------------------------------------------------------------
extern "C" void kernel_launch(void* const* d_in, const int* in_sizes, int n_in,
                              void* d_out, int out_size, void* d_ws, size_t ws_size,
                              hipStream_t stream) {
  const float* x     = (const float*)d_in[0];
  const float* Wq    = (const float*)d_in[1];
  const float* Wk    = (const float*)d_in[2];
  const float* Wv    = (const float*)d_in[3];
  const float* gamma = (const float*)d_in[4];
  float* out = (float*)d_out;

  char* ws = (char*)d_ws;
  unsigned short* WT  = (unsigned short*)(ws);                         // 640KB
  unsigned short* qws = (unsigned short*)(ws + 655360);                // 2MB
  unsigned short* kws = (unsigned short*)(ws + 655360 + 2097152);      // 2MB
  unsigned short* vT  = (unsigned short*)(ws + 655360 + 2 * 2097152);  // 16MB

  hipLaunchKernelGGL(pam_wt,   dim3(1280), dim3(256), 0, stream, Wq, Wk, Wv, WT);
  hipLaunchKernelGGL(pam_proj, dim3(256),  dim3(256), 0, stream, x, WT, qws, kws, vT);
  hipLaunchKernelGGL(pam_attn, dim3(256),  dim3(256), 0, stream, x, qws, kws, vT, gamma, out);
}
